// NTM_53240414601578
// MI455X (gfx1250) — matrix operand order repaired and checked by reading, not compiled, not run on hardware
//
#include <hip/hip_runtime.h>
#include <math.h>

// ---------------------------------------------------------------------------
// NTM forward for MI455X (gfx1250), wave32 / WGP model.
//
// Strategy (latency-bound recurrence, weights L2-resident):
//  - 16 workgroups, one per batch element, 256 threads (8 wave32) each.
//  - Whole S=256 recurrence inside one kernel; no cross-WG sync needed.
//  - WMMA f32_16x16x32_f16 for the true matrix-matrix shapes:
//      sim[h,n] = k_n[h,:64] . mem_n[n,:64]   (heads padded 4->16, 8 n-tiles)
//      rv[h,m]  = (rw*rownorm)[h,:128] . mem_n[:,m]  (4 m-tiles, K=128)
//    Fragments use contiguous k order (k = 16*(lane/16) + i) in A and B.
//  - GEMVs (gates 640->1024, head proj 256->1280, output 512->256) are
//    coalesced f32 VALU dot products over column-major repacked weights
//    (built once per call by init kernels into d_ws), with global prefetch.
//  - f32 memory state kept in an L2-resident d_ws region (32KB/batch);
//    normalized f16 copy in LDS for the WMMA paths. Static LDS ~43KB.
// ---------------------------------------------------------------------------

typedef __attribute__((ext_vector_type(16))) _Float16 v16h;
typedef __attribute__((ext_vector_type(8)))  float    v8f;

#define BB    16
#define SS    256
#define IND   128
#define OUTD  256
#define HCD   256
#define NM    128   // memory slots
#define MD    64    // memory width
#define NHD   4     // heads
#define ZD    640   // IN + H*M + HC
#define G1D   1024  // LSTM gates
#define G2D   1280  // fused head projections (1072 used, padded)

// d_ws layout (floats)
#define OFF_W1  0                               // [640][1024]
#define OFF_B1  (OFF_W1 + ZD * G1D)             // [1024]
#define OFF_W2  (OFF_B1 + G1D)                  // [256][1280]
#define OFF_B2  (OFF_W2 + HCD * G2D)            // [1280]
#define OFF_WO  (OFF_B2 + G2D)                  // [512][256]
#define OFF_MEM (OFF_WO + 512 * OUTD)           // [16][128*64]
// total = OFF_MEM + 16*8192 = 1,247,488 floats (~4.76 MB) of workspace.

__device__ __forceinline__ float sig_(float x)  { return 1.f / (1.f + expf(-x)); }
__device__ __forceinline__ float sp_(float x)   { return (x > 20.f) ? x : log1pf(expf(x)); }

// ---------------------------------------------------------------------------
// Init kernels: repack weights into GEMV-friendly [k][j] layouts in d_ws.
// ---------------------------------------------------------------------------
__global__ void init_w1(const float* __restrict__ w_ih, const float* __restrict__ w_hh,
                        const float* __restrict__ b_ih, const float* __restrict__ b_hh,
                        float* __restrict__ ws) {
  int i = blockIdx.x * 256 + threadIdx.x;
  if (i < ZD * G1D) {
    int k = i / G1D, j = i - k * G1D;
    float v = (k < 384) ? w_ih[j * 384 + k] : w_hh[j * 256 + (k - 384)];
    ws[OFF_W1 + i] = v;
  }
  if (i < G1D) ws[OFF_B1 + i] = b_ih[i] + b_hh[i];
}

__device__ float w2_elem(int c, int j,
                         const float* rkW, const float* wkW, const float* erW, const float* adW,
                         const float* rsW, const float* wsW,
                         const float* rbW, const float* rgW, const float* rgmW,
                         const float* wbW, const float* wgW, const float* wgmW) {
  if (j < 256)  { int h = j >> 6, m = j & 63;            return rkW[(h * 256 + c) * 64 + m]; }
  if (j < 512)  { int jj = j - 256;  int h = jj >> 6, m = jj & 63; return wkW[(h * 256 + c) * 64 + m]; }
  if (j < 768)  { int jj = j - 512;  int h = jj >> 6, m = jj & 63; return erW[(h * 256 + c) * 64 + m]; }
  if (j < 1024) { int jj = j - 768;  int h = jj >> 6, m = jj & 63; return adW[(h * 256 + c) * 64 + m]; }
  if (j < 1036) { int jj = j - 1024; int h = jj / 3, s = jj % 3;   return rsW[(h * 256 + c) * 3 + s]; }
  if (j < 1048) { int jj = j - 1036; int h = jj / 3, s = jj % 3;   return wsW[(h * 256 + c) * 3 + s]; }
  if (j < 1052) return rbW [(j - 1048) * 256 + c];
  if (j < 1056) return rgW [(j - 1052) * 256 + c];
  if (j < 1060) return rgmW[(j - 1056) * 256 + c];
  if (j < 1064) return wbW [(j - 1060) * 256 + c];
  if (j < 1068) return wgW [(j - 1064) * 256 + c];
  if (j < 1072) return wgmW[(j - 1068) * 256 + c];
  return 0.f;
}

__device__ float b2_elem(int j,
                         const float* rkb, const float* wkb, const float* erb, const float* adb,
                         const float* rsb, const float* wsb,
                         const float* rbb, const float* rgb, const float* rgmb,
                         const float* wbb, const float* wgb, const float* wgmb) {
  if (j < 256)  return rkb[j];
  if (j < 512)  return wkb[j - 256];
  if (j < 768)  return erb[j - 512];
  if (j < 1024) return adb[j - 768];
  if (j < 1036) return rsb[j - 1024];
  if (j < 1048) return wsb[j - 1036];
  if (j < 1052) return rbb [j - 1048];
  if (j < 1056) return rgb [j - 1052];
  if (j < 1060) return rgmb[j - 1056];
  if (j < 1064) return wbb [j - 1060];
  if (j < 1068) return wgb [j - 1064];
  if (j < 1072) return wgmb[j - 1068];
  return 0.f;
}

__global__ void init_w2(const float* rkW, const float* wkW, const float* erW, const float* adW,
                        const float* rsW, const float* wsW,
                        const float* rbW, const float* rgW, const float* rgmW,
                        const float* wbW, const float* wgW, const float* wgmW,
                        const float* rkb, const float* wkb, const float* erb, const float* adb,
                        const float* rsb, const float* wsb,
                        const float* rbb, const float* rgb, const float* rgmb,
                        const float* wbb, const float* wgb, const float* wgmb,
                        float* __restrict__ ws) {
  int i = blockIdx.x * 256 + threadIdx.x;
  if (i < HCD * G2D) {
    int c = i / G2D, j = i - c * G2D;
    ws[OFF_W2 + i] = w2_elem(c, j, rkW, wkW, erW, adW, rsW, wsW, rbW, rgW, rgmW, wbW, wgW, wgmW);
  }
  if (i < G2D)
    ws[OFF_B2 + i] = b2_elem(i, rkb, wkb, erb, adb, rsb, wsb, rbb, rgb, rgmb, wbb, wgb, wgmb);
}

__global__ void init_wo(const float* __restrict__ out_W, float* __restrict__ ws) {
  int i = blockIdx.x * 256 + threadIdx.x;
  if (i < 512 * OUTD) {
    int k = i / OUTD, j = i - k * OUTD;
    ws[OFF_WO + i] = out_W[j * 512 + k];
  }
}

// ---------------------------------------------------------------------------
// Main recurrence kernel: one workgroup per batch element.
// ---------------------------------------------------------------------------
__launch_bounds__(256, 1)
__global__ void ntm_main(const int* __restrict__ x,
                         const float* __restrict__ embedding,
                         const float* __restrict__ init_mem,
                         const float* __restrict__ out_b,
                         float* __restrict__ ws,
                         float* __restrict__ out) {
  const int tid = threadIdx.x;
  const int wv  = tid >> 5;      // wave id 0..7
  const int ln  = tid & 31;      // lane id
  const int b   = blockIdx.x;

  const float* W1  = ws + OFF_W1;
  const float* Bi1 = ws + OFF_B1;
  const float* W2  = ws + OFF_W2;
  const float* Bi2 = ws + OFF_B2;
  const float* Wo  = ws + OFF_WO;
  float*       gmem = ws + OFF_MEM + (size_t)b * (NM * MD);   // f32 memory state (L2)

  __shared__ float s_z[ZD];               // LSTM input / output concat buffer
  __shared__ float s_g[G2D];              // gate / projection outputs (shared buffer)
  __shared__ float s_h[HCD], s_c[HCD];
  __shared__ float s_rw[NHD][NM], s_ww[NHD][NM];
  __shared__ float s_rv[NHD][MD];
  __shared__ float s_ev[NHD][MD], s_av[NHD][MD];
  __shared__ float s_sim[NHD][132];
  __shared__ float s_wgt[NHD][132];
  __shared__ float s_rownorm[NM], s_rowinv[NM];
  __shared__ float s_kinv[NHD];
  __shared__ float s_scal[24];            // rb,rg,rgm,wb,wg,wgm (activated)
  __shared__ float s_shift[8][3];         // softmaxed shift (read 0..3, write 4..7)
  __shared__ _Float16 s_memn16[NM][66];   // normalized memory, f16 (WMMA B operand)
  __shared__ _Float16 s_A16[16][132];     // WMMA A operand staging (k_n or rw*norm)

  // ---- per-launch state init (deterministic across graph replays) ----
  s_h[tid] = 0.f; s_c[tid] = 0.f;
  #pragma unroll
  for (int e = 0; e < 2; ++e) {
    int lin = tid + e * 256;                       // 512 = 4*128
    s_rw[lin >> 7][lin & 127] = 0.f;
    s_ww[lin >> 7][lin & 127] = 0.f;
  }
  s_rv[tid >> 6][tid & 63] = 0.f;
  #pragma unroll
  for (int e = 0; e < 32; ++e) {
    int idx = tid + e * 256;
    gmem[idx] = init_mem[idx];                     // broadcast initial memory
  }
  __threadfence_block();
  __syncthreads();

  #pragma unroll 1
  for (int t = 0; t < SS; ++t) {
    // ---------------- build z = [emb, rv, h_prev] ----------------
    const int xi = x[b * SS + t];
    if (tid < IND) s_z[tid] = embedding[xi * IND + tid];
    s_z[IND + tid]       = s_rv[tid >> 6][tid & 63];
    s_z[IND + 256 + tid] = s_h[tid];
    __syncthreads();

    // ---------------- GEMV1: gates = W1^T z + (b_ih+b_hh) ----------------
    {
      float4 acc = ((const float4*)(Bi1))[tid];
      #pragma unroll 1
      for (int k = 0; k < ZD; k += 4) {
        __builtin_prefetch(W1 + (size_t)(k + 32) * G1D + (tid << 2), 0, 1);
        #pragma unroll
        for (int kk = 0; kk < 4; ++kk) {
          float zk = s_z[k + kk];
          float4 w = ((const float4*)(W1 + (size_t)(k + kk) * G1D))[tid];
          acc.x += w.x * zk; acc.y += w.y * zk; acc.z += w.z * zk; acc.w += w.w * zk;
        }
      }
      int j = tid << 2;
      s_g[j] = acc.x; s_g[j + 1] = acc.y; s_g[j + 2] = acc.z; s_g[j + 3] = acc.w;
    }
    __syncthreads();

    // ---------------- LSTM cell (torch gate order i,f,g,o) ----------------
    {
      float gi = s_g[tid], gf = s_g[256 + tid], gg = s_g[512 + tid], go = s_g[768 + tid];
      float cn = sig_(gf) * s_c[tid] + sig_(gi) * tanhf(gg);
      float hn = sig_(go) * tanhf(cn);
      __syncthreads();              // all gate reads done before s_g reuse
      s_c[tid] = cn; s_h[tid] = hn;
    }
    __syncthreads();

    // ---------------- GEMV2: fused head projections from h_new ----------------
    {
      float acc[5];
      #pragma unroll
      for (int u = 0; u < 5; ++u) acc[u] = Bi2[tid + u * 256];
      #pragma unroll 1
      for (int c2 = 0; c2 < HCD; ++c2) {
        float hv = s_h[c2];
        const float* row = W2 + (size_t)c2 * G2D;
        #pragma unroll
        for (int u = 0; u < 5; ++u) acc[u] += row[tid + u * 256] * hv;
      }
      #pragma unroll
      for (int u = 0; u < 5; ++u) s_g[tid + u * 256] = acc[u];
    }
    __syncthreads();

    // ---------------- activations: scalars, shifts, erase/add vectors ----------------
    if (tid < 24) {
      float raw = s_g[1048 + tid];
      int grp = tid >> 2;                          // 0 rb,1 rg,2 rgm,3 wb,4 wg,5 wgm
      float v;
      if (grp == 1 || grp == 4) v = sig_(raw);
      else { v = sp_(raw); if (grp == 2 || grp == 5) v += 1.f; }
      s_scal[tid] = v;
    }
    if (tid < 8) {
      int base = (tid < 4) ? (1024 + tid * 3) : (1036 + (tid - 4) * 3);
      float a0 = s_g[base], a1 = s_g[base + 1], a2 = s_g[base + 2];
      float mx = fmaxf(a0, fmaxf(a1, a2));
      float e0 = expf(a0 - mx), e1 = expf(a1 - mx), e2 = expf(a2 - mx);
      float inv = 1.f / (e0 + e1 + e2);
      s_shift[tid][0] = e0 * inv; s_shift[tid][1] = e1 * inv; s_shift[tid][2] = e2 * inv;
    }
    {
      int h = tid >> 6, m = tid & 63;
      s_ev[h][m] = sig_(s_g[512 + tid]);
      s_av[h][m] = tanhf(s_g[768 + tid]);
    }
    __syncthreads();

    // ---------------- memory row norms + normalized f16 copy ----------------
    if (tid < NM) {
      const float* row = gmem + tid * MD;
      float ss = 0.f;
      #pragma unroll 1
      for (int m = 0; m < MD; ++m) { float v = row[m]; ss += v * v; }
      float sc = fmaxf(sqrtf(ss), 1e-12f);
      s_rownorm[tid] = sc;
      s_rowinv[tid]  = 1.f / sc;
    }
    __syncthreads();
    #pragma unroll
    for (int e = 0; e < 32; ++e) {
      int idx = tid + e * 256;
      int n = idx >> 6, m = idx & 63;
      s_memn16[n][m] = (_Float16)(gmem[idx] * s_rowinv[n]);
    }
    __syncthreads();

    // ---------------- addressing: phase 0 = read, phase 1 = write ----------------
    #pragma unroll 1
    for (int phase = 0; phase < 2; ++phase) {
      const int koff  = phase ? 256 : 0;           // wk vs rk offset in s_g
      const int sbase = phase ? 12 : 0;            // scalar block offset
      const int shb   = phase ? 4 : 0;             // shift block offset
      float* prevW = phase ? &s_ww[0][0] : &s_rw[0][0];

      // key norms (4 heads)
      if (tid < NHD) {
        float ss = 0.f;
        const int ko = koff + tid * MD;
        #pragma unroll 1
        for (int m = 0; m < MD; ++m) { float v = s_g[ko + m]; ss += v * v; }
        s_kinv[tid] = 1.f / fmaxf(sqrtf(ss), 1e-12f);
      }
      __syncthreads();

      // stage normalized keys as WMMA A operand (16x64, heads 4..15 zero)
      #pragma unroll
      for (int e = 0; e < 4; ++e) {
        int lin = tid + e * 256;                   // 1024 = 16*64
        int r = lin >> 6, m = lin & 63;
        s_A16[r][m] = (r < NHD) ? (_Float16)(s_g[koff + r * MD + m] * s_kinv[r])
                                : (_Float16)0.f;
      }
      __syncthreads();

      // sim[h, n] via WMMA: 8 waves x one 16-column n-tile, K=64 (2 steps of 32)
      {
        const int n0 = wv * 16;
        v8f acc = {0.f, 0.f, 0.f, 0.f, 0.f, 0.f, 0.f, 0.f};
        #pragma unroll
        for (int kc = 0; kc < 2; ++kc) {
          v16h a, bb;
          const int r  = ln & 15;
          const int ko = (ln >> 4) * 16;
          #pragma unroll
          for (int i = 0; i < 16; ++i) {
            int kk = ko + i;
            a[i] = s_A16[r][kc * 32 + kk];
          }
          const int nc = ln & 15;
          const int kb = (ln >> 4) * 16;
          #pragma unroll
          for (int i = 0; i < 16; ++i)
            bb[i] = s_memn16[n0 + nc][kc * 32 + kb + i];
          acc = __builtin_amdgcn_wmma_f32_16x16x32_f16(false, a, false, bb,
                                                       (short)0, acc, false, false);
        }
        if ((ln >> 4) == 0) {                      // rows 0..3 live in lanes 0..15
          #pragma unroll
          for (int r = 0; r < NHD; ++r) s_sim[r][n0 + (ln & 15)] = acc[r];
        }
      }
      __syncthreads();

      // content softmax + interpolation gate (wave h handles head h)
      if (wv < NHD) {
        const int h = wv;
        const float beta = s_scal[sbase + h];
        const float gg   = s_scal[sbase + 4 + h];
        float v[4]; float mx = -3.4e38f;
        #pragma unroll
        for (int q = 0; q < 4; ++q) { v[q] = beta * s_sim[h][ln + 32 * q]; mx = fmaxf(mx, v[q]); }
        for (int off = 16; off; off >>= 1) mx = fmaxf(mx, __shfl_xor(mx, off, 32));
        float sm = 0.f;
        #pragma unroll
        for (int q = 0; q < 4; ++q) { v[q] = expf(v[q] - mx); sm += v[q]; }
        for (int off = 16; off; off >>= 1) sm += __shfl_xor(sm, off, 32);
        const float inv = 1.f / sm;
        #pragma unroll
        for (int q = 0; q < 4; ++q) {
          int n = ln + 32 * q;
          s_wgt[h][n] = gg * v[q] * inv + (1.f - gg) * prevW[h * NM + n];
        }
      }
      __syncthreads();

      // circular conv + sharpen + renormalize
      if (wv < NHD) {
        const int h = wv;
        const float gam = s_scal[sbase + 8 + h];
        const float sh0 = s_shift[shb + h][0], sh1 = s_shift[shb + h][1], sh2 = s_shift[shb + h][2];
        float w[4]; float ps = 0.f;
        #pragma unroll
        for (int q = 0; q < 4; ++q) {
          int n = ln + 32 * q;
          float cv = sh0 * s_wgt[h][(n + 1) & 127] + sh1 * s_wgt[h][n]
                   + sh2 * s_wgt[h][(n - 1) & 127];
          cv = fmaxf(cv, 0.f);
          w[q] = powf(cv, gam);
          ps += w[q];
        }
        for (int off = 16; off; off >>= 1) ps += __shfl_xor(ps, off, 32);
        const float inv = 1.f / ps;
        #pragma unroll
        for (int q = 0; q < 4; ++q) prevW[h * NM + (ln + 32 * q)] = w[q] * inv;
      }
      __syncthreads();

      if (phase == 0) {
        // stage (rw * rownorm) as A operand (16x128): rv = A . memn16
        #pragma unroll
        for (int e = 0; e < 8; ++e) {
          int lin = tid + e * 256;                 // 2048 = 16*128
          int r = lin >> 7, n = lin & 127;
          s_A16[r][n] = (r < NHD) ? (_Float16)(s_rw[r][n] * s_rownorm[n])
                                  : (_Float16)0.f;
        }
        __syncthreads();

        // rv via WMMA: waves 0..3 each own a 16-column m-tile, K=128 (4 steps)
        if (wv < 4) {
          const int m0 = wv * 16;
          v8f acc = {0.f, 0.f, 0.f, 0.f, 0.f, 0.f, 0.f, 0.f};
          #pragma unroll
          for (int kc = 0; kc < 4; ++kc) {
            v16h a, bb;
            const int r  = ln & 15;
            const int ko = (ln >> 4) * 16;
            #pragma unroll
            for (int i = 0; i < 16; ++i) {
              int kk = ko + i;
              a[i] = s_A16[r][kc * 32 + kk];
            }
            const int nc = ln & 15;
            const int kb = (ln >> 4) * 16;
            #pragma unroll
            for (int i = 0; i < 16; ++i)
              bb[i] = s_memn16[kc * 32 + kb + i][m0 + nc];
            acc = __builtin_amdgcn_wmma_f32_16x16x32_f16(false, a, false, bb,
                                                         (short)0, acc, false, false);
          }
          if ((ln >> 4) == 0) {
            #pragma unroll
            for (int r = 0; r < NHD; ++r) s_rv[r][m0 + (ln & 15)] = acc[r];
          }
        }
        __syncthreads();
      }
    } // phases

    // ---------------- memory update: mem = mem*(1 - ww^T ev) + ww^T av ----------------
    #pragma unroll
    for (int e = 0; e < 32; ++e) {
      int idx = tid + e * 256;
      int n = idx >> 6, m = idx & 63;
      float er = 0.f, ad = 0.f;
      #pragma unroll
      for (int hh = 0; hh < NHD; ++hh) {
        float wn = s_ww[hh][n];
        er += wn * s_ev[hh][m];
        ad += wn * s_av[hh][m];
      }
      gmem[idx] = gmem[idx] * (1.f - er) + ad;
    }
    __threadfence_block();
    __syncthreads();

    // ---------------- output logits = Wo^T [h, rv] + out_b ----------------
    s_z[tid]       = s_h[tid];
    s_z[256 + tid] = s_rv[tid >> 6][tid & 63];
    __syncthreads();
    {
      float acc = out_b[tid];
      #pragma unroll 1
      for (int k = 0; k < 512; ++k) acc += Wo[(size_t)k * OUTD + tid] * s_z[k];
      out[((size_t)b * SS + t) * OUTD + tid] = acc;
    }
    __syncthreads();
  }
}

// ---------------------------------------------------------------------------
extern "C" void kernel_launch(void* const* d_in, const int* in_sizes, int n_in,
                              void* d_out, int out_size, void* d_ws, size_t ws_size,
                              hipStream_t stream) {
  (void)in_sizes; (void)n_in; (void)out_size; (void)ws_size;
  const int*   x    = (const int*)  d_in[0];
  const float* emb  = (const float*)d_in[1];
  const float* im   = (const float*)d_in[2];
  const float* w_ih = (const float*)d_in[3];
  const float* w_hh = (const float*)d_in[4];
  const float* b_ih = (const float*)d_in[5];
  const float* b_hh = (const float*)d_in[6];
  const float* rkW  = (const float*)d_in[7];   const float* rkb  = (const float*)d_in[8];
  const float* rbW  = (const float*)d_in[9];   const float* rbb  = (const float*)d_in[10];
  const float* rgW  = (const float*)d_in[11];  const float* rgb  = (const float*)d_in[12];
  const float* rsW  = (const float*)d_in[13];  const float* rsb  = (const float*)d_in[14];
  const float* rgmW = (const float*)d_in[15];  const float* rgmb = (const float*)d_in[16];
  const float* wkW  = (const float*)d_in[17];  const float* wkb  = (const float*)d_in[18];
  const float* wbW  = (const float*)d_in[19];  const float* wbb  = (const float*)d_in[20];
  const float* wgW  = (const float*)d_in[21];  const float* wgb  = (const float*)d_in[22];
  const float* wsW  = (const float*)d_in[23];  const float* wsb  = (const float*)d_in[24];
  const float* wgmW = (const float*)d_in[25];  const float* wgmb = (const float*)d_in[26];
  const float* erW  = (const float*)d_in[27];  const float* erb  = (const float*)d_in[28];
  const float* adW  = (const float*)d_in[29];  const float* adb  = (const float*)d_in[30];
  const float* outW = (const float*)d_in[31];  const float* outb = (const float*)d_in[32];

  float* ws  = (float*)d_ws;
  float* out = (float*)d_out;

  init_w1<<<(ZD * G1D + 255) / 256, 256, 0, stream>>>(w_ih, w_hh, b_ih, b_hh, ws);
  init_w2<<<(HCD * G2D + 255) / 256, 256, 0, stream>>>(
      rkW, wkW, erW, adW, rsW, wsW, rbW, rgW, rgmW, wbW, wgW, wgmW,
      rkb, wkb, erb, adb, rsb, wsb, rbb, rgb, rgmb, wbb, wgb, wgmb, ws);
  init_wo<<<(512 * OUTD + 255) / 256, 256, 0, stream>>>(outW, ws);

  ntm_main<<<BB, 256, 0, stream>>>(x, emb, im, outb, ws, out);
}
